// RNNTransducer_75471165325995
// MI455X (gfx1250) — matrix operand order repaired and checked by reading, not compiled, not run on hardware
//
// RNN-Transducer for MI455X (gfx1250, wave32, WMMA).
//  - Joint split linearly: tanh((e+g)W+b) = tanh(eW + gW + b): 17.2 G-MAC -> 0.34 G-MAC
//    + one HBM-bound 134MB pass (~6us at 23.3 TB/s).
//  - All matmuls via v_wmma_f32_16x16x32_bf16 (8x the K-depth of fp32 WMMA),
//    fp32 accumulation and fp32 cell state.
//  - Weights repacked once to lane-major bf16 B-tiles (2x global_load_b128/frag);
//    activations pre-converted to padded bf16 so A-frags are 2x b128 loads, no guards.
//  - Sequential LSTM chains: encoder block 0 || decoder block 1; 32 waves each;
//    M=8 batch in one padded 16-row tile; per-wave gate-column ownership keeps
//    i/j/f/o of each cell column in the same wave's accumulators; c in registers.
//  - tanh/sigmoid use the CDNA5 hardware V_TANH_F32 TRANS op (sig(x)=.5*tanh(x/2)+.5),
//    keeping the 33.5M-element joint pass HBM-bound instead of VALU-bound.
#include <hip/hip_runtime.h>
#include <hip/hip_bf16.h>

typedef __attribute__((ext_vector_type(16))) __bf16 v16bf;
typedef __attribute__((ext_vector_type(8)))  float  v8f;

union Frag32 {
  v16bf v;
  uint4 q[2];
  unsigned short s[16];
};

__device__ __forceinline__ unsigned short f2bf(float x) {
  unsigned int b = __float_as_uint(x);
  b += 0x7FFFu + ((b >> 16) & 1u);          // round-to-nearest-even
  return (unsigned short)(b >> 16);
}

// CDNA5 hardware tanh (TRANS32). One op vs ~20+ VALU for libm tanhf.
__device__ __forceinline__ float htanh(float x) {
  float y;
  asm("v_tanh_f32 %0, %1" : "=v"(y) : "v"(x));
  return y;
}
__device__ __forceinline__ float sigf(float x) {  // exact: 1/(1+e^-x)
  return __builtin_fmaf(0.5f, htanh(0.5f * x), 0.5f);
}

__device__ __forceinline__ v8f wmma_bf16(v16bf a, v16bf b, v8f c) {
  return __builtin_amdgcn_wmma_f32_16x16x32_bf16(false, a, false, b, (short)0, c,
                                                 false, false);
}

// B fragment: packed tile is lane-major, 16 bf16 (32B) per lane.
__device__ __forceinline__ v16bf load_bfrag(const unsigned short* __restrict__ Bp,
                                            long long tile) {
  int L = threadIdx.x & 31;
  const uint4* p = (const uint4*)(Bp + tile * 512 + (long long)L * 16);
  Frag32 f;
  f.q[0] = p[0];
  f.q[1] = p[1];
  return f.v;
}

// A fragment (16x32 bf16) from a padded bf16 buffer (global or LDS), rows 0..15
// all readable. Lane L -> row (L&15); lanes<16: K runs [0..7],[16..23];
// lanes>=16: [8..15],[24..31]. Two unconditional 128-bit loads.
__device__ __forceinline__ v16bf load_afrag(const unsigned short* base, int stride,
                                            int kt) {
  int L = threadIdx.x & 31;
  const unsigned short* p =
      base + (long long)(L & 15) * stride + kt * 32 + ((L >> 4) << 3);
  Frag32 f;
  f.q[0] = *(const uint4*)p;
  f.q[1] = *(const uint4*)(p + 16);
  return f.v;
}

// ---- weight repack: fp32 [Ksrc,N] row-major -> bf16 WMMA B tiles (one-time)
__global__ void pack_w_kernel(const float* __restrict__ W, int Ksrc, int N,
                              unsigned short* __restrict__ dst,
                              int ktTotal, int ktStart, int ktCount) {
  long long e = (long long)blockIdx.x * blockDim.x + threadIdx.x;
  long long total = (long long)ktCount * (N >> 4) * 512;
  if (e >= total) return;
  int within = (int)(e & 511);
  long long tile = e >> 9;
  int ktL = (int)(tile % ktCount);
  long long nt = tile / ktCount;
  int Lp = within >> 4;
  int s  = within & 15;
  int k  = ktL * 32 + ((Lp >= 16) ? 8 : 0) + ((s >= 8) ? 8 : 0) + s;
  int n  = (int)nt * 16 + (Lp & 15);
  float x = (k < Ksrc) ? W[(long long)k * N + n] : 0.f;
  dst[(nt * ktTotal + ktStart + ktL) * 512 + within] = f2bf(x);
}

// ---- activation convert: fp32 [M,K] -> bf16 [M,Kpad] (K zero-padded)
__global__ void cvt_pad_kernel(const float* __restrict__ src, int K,
                               unsigned short* __restrict__ dst, int Kpad,
                               long long total) {
  long long i = (long long)blockIdx.x * blockDim.x + threadIdx.x;
  if (i >= total) return;
  int k = (int)(i % Kpad);
  long long row = i / Kpad;
  float x = (k < K) ? src[row * K + k] : 0.f;
  dst[i] = f2bf(x);
}

// ---- parallel GEMM: C[M,N] = Abf16[M,Kpad] * Bpacked + bias, M%16==0.
// One 16x16 tile per wave. tshift>=0: row = b*(1<<tshift)+t stored at
// C[(t*8+b)*N+col]; tshift<0: row-major.
__global__ void gemm_bf_kernel(const unsigned short* __restrict__ Abf, int Kpad,
                               const unsigned short* __restrict__ Bp, int KT, int N,
                               const float* __restrict__ bias,
                               float* __restrict__ C, int M, int tshift) {
  int wave = blockIdx.x * (blockDim.x >> 5) + (threadIdx.x >> 5);
  int ntT = N >> 4;
  if (wave >= (M >> 4) * ntT) return;
  int mt = wave / ntT, nt = wave % ntT;
  int L = threadIdx.x & 31;
  int lane = L & 15;

  float binit = bias ? bias[nt * 16 + lane] : 0.f;
  v8f acc;
#pragma unroll
  for (int r = 0; r < 8; ++r) acc[r] = binit;

  const unsigned short* Abase = Abf + (long long)mt * 16 * Kpad;
  for (int kt = 0; kt < KT; ++kt) {
    v16bf va = load_afrag(Abase, Kpad, kt);
    v16bf vb = load_bfrag(Bp, (long long)nt * KT + kt);
    acc = wmma_bf16(va, vb, acc);
  }

  int rbase = (L >= 16) ? 8 : 0;          // C/D: lanes>=16 hold rows 8..15
  int col = nt * 16 + lane;
#pragma unroll
  for (int r = 0; r < 8; ++r) {
    int row = mt * 16 + rbase + r;
    long long o;
    if (tshift >= 0) {
      int b = row >> tshift;
      int t = row & ((1 << tshift) - 1);
      o = ((long long)t * 8 + b) * N + col;
    } else {
      o = (long long)row * N + col;
    }
    C[o] = acc[r];
  }
}

// ---- sequential 2-layer projected-LSTM chain (one block per chain)
struct ChainArgs {
  const float* xgT;               // [T][8][4096] = x@Wx0 + b0 (fp32)
  const unsigned short* Wh0p;     // K=512  -> 16 kt, N=4096
  const unsigned short* Wp0p;     // K=1024 -> 32 kt, N=512
  unsigned short* e0;             // bf16 [8,T,512] layer-0 output
  const unsigned short* W1p;      // stacked [Wx1;Wh1] K=1024 -> 32 kt, N=4096
  const float* b1;
  const unsigned short* Wp1p;
  unsigned short* e1;             // bf16 [8,T,512] layer-1 output
  int T;
};

__global__ __launch_bounds__(1024) void lstm_chain_kernel(ChainArgs a0, ChainArgs a1) {
  ChainArgs A = (blockIdx.x == 0) ? a0 : a1;
  __shared__ __align__(16) unsigned short Abuf[16 * 1024];  // bf16 A rows (rows 8-15 = 0)
  __shared__ __align__(16) unsigned short Hbuf[16 * 1024];  // bf16 h (rows 8-15 = 0)

  int tid = threadIdx.x;
  int w = tid >> 5;                // wave 0..31
  int L = tid & 31;
  int lane = L & 15;

  for (int i = tid; i < 16 * 1024; i += 1024) Hbuf[i] = 0;

  for (int layer = 0; layer < 2; ++layer) {
    const unsigned short* Wg = layer ? A.W1p : A.Wh0p;
    const unsigned short* Wp = layer ? A.Wp1p : A.Wp0p;
    const unsigned short* inSeq = layer ? A.e0 : nullptr;
    unsigned short* outSeq = layer ? A.e1 : A.e0;
    int KTg  = layer ? 32 : 16;
    int mOff = layer ? 512 : 0;

    for (int i = tid; i < 16 * 1024; i += 1024) Abuf[i] = 0;  // m0=0 + zero pad rows
    float creg[16];
#pragma unroll
    for (int i = 0; i < 16; ++i) creg[i] = 0.f;
    __syncthreads();

    for (int t = 0; t < A.T; ++t) {
      if (layer) {  // stage e0[t] (already bf16) into Abuf K[0..511]
        for (int i = tid; i < 8 * 512; i += 1024) {
          int b = i >> 9, k = i & 511;
          Abuf[b * 1024 + k] = inSeq[((long long)b * A.T + t) * 512 + k];
        }
      }
      __syncthreads();

      // ---- gate GEMM: g[8,4096] = [in|m]@Wg  (+ xg[t] or b1 as C-init)
      v8f acc[8];
#pragma unroll
      for (int g = 0; g < 4; ++g) {
#pragma unroll
        for (int j = 0; j < 2; ++j) {
          int col = (g * 64 + w * 2 + j) * 16 + lane;
          v8f ai;
          if (!layer) {
            if (L < 16) {
              const float* xp = A.xgT + (long long)t * 8 * 4096 + col;
#pragma unroll
              for (int r = 0; r < 8; ++r) ai[r] = xp[(long long)r * 4096];
            } else {
#pragma unroll
              for (int r = 0; r < 8; ++r) ai[r] = 0.f;
            }
          } else {
            float bv = A.b1[col];
#pragma unroll
            for (int r = 0; r < 8; ++r) ai[r] = bv;
          }
          acc[g * 2 + j] = ai;
        }
      }
      for (int kt = 0; kt < KTg; ++kt) {
        v16bf va = load_afrag(Abuf, 1024, kt);
#pragma unroll
        for (int g = 0; g < 4; ++g) {
#pragma unroll
          for (int j = 0; j < 2; ++j) {
            int tileN = g * 64 + w * 2 + j;
            v16bf vb = load_bfrag(Wg, (long long)tileN * KTg + kt);
            acc[g * 2 + j] = wmma_bf16(va, vb, acc[g * 2 + j]);
          }
        }
      }

      // ---- elementwise LSTM cell (TF gate order i,j,f,o; forget bias 1.0)
      if (L < 16) {
#pragma unroll
        for (int j = 0; j < 2; ++j) {
          int col = w * 32 + j * 16 + lane;     // cell index 0..1023
#pragma unroll
          for (int r = 0; r < 8; ++r) {
            float iv = acc[0 + j][r];
            float jv = acc[2 + j][r];
            float fv = acc[4 + j][r];
            float ov = acc[6 + j][r];
            float c = sigf(fv + 1.f) * creg[j * 8 + r] + sigf(iv) * htanh(jv);
            creg[j * 8 + r] = c;
            float h = sigf(ov) * htanh(c);
            Hbuf[r * 1024 + col] = f2bf(h);
          }
        }
      }
      __syncthreads();

      // ---- projection: m[8,512] = h[8,1024] @ Wp ; wave w -> N-tile w
      v8f pacc;
#pragma unroll
      for (int r = 0; r < 8; ++r) pacc[r] = 0.f;
      for (int kt = 0; kt < 32; ++kt) {
        v16bf va = load_afrag(Hbuf, 1024, kt);
        v16bf vb = load_bfrag(Wp, (long long)w * 32 + kt);
        pacc = wmma_bf16(va, vb, pacc);
      }
      if (L < 16) {
        int col = w * 16 + lane;
#pragma unroll
        for (int r = 0; r < 8; ++r) {
          unsigned short mb = f2bf(pacc[r]);
          outSeq[((long long)r * A.T + t) * 512 + col] = mb;   // bf16 [B,T,P]
          Abuf[r * 1024 + mOff + col] = mb;                    // next-step m
        }
      }
      __syncthreads();
    }
    __syncthreads();
  }
}

// ---- fused joint: out = tanh(E'[b,t,v] + G'[b,u,v])  (bias folded into E')
__global__ void joint_final_kernel(const float* __restrict__ Ep,
                                   const float* __restrict__ Gp,
                                   float* __restrict__ out,
                                   int T, int U, long long total) {
  long long i = ((long long)blockIdx.x * blockDim.x + threadIdx.x) * 4;
  if (i >= total) return;
  int v = (int)(i & 255);
  long long r = i >> 8;                 // (b*T+t)*U + u
  int u = (int)(r % U);
  long long bt = r / U;
  int b = (int)(bt / T);
  const float4 e = *(const float4*)(Ep + bt * 256 + v);
  const float4 g = *(const float4*)(Gp + ((long long)(b * U + u)) * 256 + v);
  float4 o;
  o.x = htanh(e.x + g.x);
  o.y = htanh(e.y + g.y);
  o.z = htanh(e.z + g.z);
  o.w = htanh(e.w + g.w);
  *(float4*)(out + i) = o;
}

extern "C" void kernel_launch(void* const* d_in, const int* in_sizes, int n_in,
                              void* d_out, int out_size, void* d_ws, size_t ws_size,
                              hipStream_t stream) {
  (void)in_sizes; (void)n_in; (void)out_size; (void)ws_size;
  const float* x        = (const float*)d_in[0];   // [8,256,80]
  const float* prev     = (const float*)d_in[1];   // [8,64,256]
  const float* enc_Wx0  = (const float*)d_in[2];
  const float* enc_Wh0  = (const float*)d_in[3];
  const float* enc_b0   = (const float*)d_in[4];
  const float* enc_Wp0  = (const float*)d_in[5];
  const float* enc_Wx1  = (const float*)d_in[6];
  const float* enc_Wh1  = (const float*)d_in[7];
  const float* enc_b1   = (const float*)d_in[8];
  const float* enc_Wp1  = (const float*)d_in[9];
  const float* dec_Wx0  = (const float*)d_in[10];
  const float* dec_Wh0  = (const float*)d_in[11];
  const float* dec_b0   = (const float*)d_in[12];
  const float* dec_Wp0  = (const float*)d_in[13];
  const float* dec_Wx1  = (const float*)d_in[14];
  const float* dec_Wh1  = (const float*)d_in[15];
  const float* dec_b1   = (const float*)d_in[16];
  const float* dec_Wp1  = (const float*)d_in[17];
  const float* joint_W  = (const float*)d_in[18];
  const float* joint_b  = (const float*)d_in[19];
  float* out = (float*)d_out;

  const int T = 256, U = 64;           // tshift 8 and 6

  char* ws = (char*)d_ws;
  size_t off = 0;
  auto take = [&](size_t bytes) -> void* {
    void* p = ws + off;
    off = (off + bytes + 255) & ~(size_t)255;
    return p;
  };
  unsigned short* p_encWx0 = (unsigned short*)take((size_t)3  * 256 * 1024);
  unsigned short* p_encWh0 = (unsigned short*)take((size_t)16 * 256 * 1024);
  unsigned short* p_encW1  = (unsigned short*)take((size_t)32 * 256 * 1024);
  unsigned short* p_encWp0 = (unsigned short*)take((size_t)32 * 32 * 1024);
  unsigned short* p_encWp1 = (unsigned short*)take((size_t)32 * 32 * 1024);
  unsigned short* p_decWx0 = (unsigned short*)take((size_t)8  * 256 * 1024);
  unsigned short* p_decWh0 = (unsigned short*)take((size_t)16 * 256 * 1024);
  unsigned short* p_decW1  = (unsigned short*)take((size_t)32 * 256 * 1024);
  unsigned short* p_decWp0 = (unsigned short*)take((size_t)32 * 32 * 1024);
  unsigned short* p_decWp1 = (unsigned short*)take((size_t)32 * 32 * 1024);
  unsigned short* p_jW     = (unsigned short*)take((size_t)16 * 16 * 1024);
  unsigned short* xbf      = (unsigned short*)take((size_t)8 * T * 96 * 2);   // x pad K80->96
  unsigned short* pbf      = (unsigned short*)take((size_t)8 * U * 256 * 2);  // prev
  float* xgT_enc = (float*)take((size_t)T * 8 * 4096 * 4);
  float* xgT_dec = (float*)take((size_t)U * 8 * 4096 * 4);
  unsigned short* e0_enc = (unsigned short*)take((size_t)8 * T * 512 * 2);
  unsigned short* e1_enc = (unsigned short*)take((size_t)8 * T * 512 * 2);
  unsigned short* e0_dec = (unsigned short*)take((size_t)8 * U * 512 * 2);
  unsigned short* e1_dec = (unsigned short*)take((size_t)8 * U * 512 * 2);
  float* Ep = (float*)take((size_t)8 * T * 256 * 4);
  float* Gp = (float*)take((size_t)8 * U * 256 * 4);

  auto pack = [&](const float* W, int K, int N, unsigned short* dst,
                  int ktTotal, int ktStart, int ktCount) {
    long long total = (long long)ktCount * (N / 16) * 512;
    pack_w_kernel<<<(int)((total + 255) / 256), 256, 0, stream>>>(
        W, K, N, dst, ktTotal, ktStart, ktCount);
  };
  auto cvt = [&](const float* src, int M, int K, unsigned short* dst, int Kpad) {
    long long total = (long long)M * Kpad;
    cvt_pad_kernel<<<(int)((total + 255) / 256), 256, 0, stream>>>(src, K, dst, Kpad,
                                                                   total);
  };
  auto gemm = [&](const unsigned short* Abf, int Kpad, int M,
                  const unsigned short* Bp, int KT, int N,
                  const float* bias, float* C, int tshift) {
    int waves = (M / 16) * (N / 16);
    gemm_bf_kernel<<<(waves + 7) / 8, 256, 0, stream>>>(Abf, Kpad, Bp, KT, N, bias, C,
                                                        M, tshift);
  };

  // ---- repack weights to bf16 WMMA tiles (one-time; L2-resident afterwards)
  pack(enc_Wx0,  80, 4096, p_encWx0, 3, 0, 3);
  pack(enc_Wh0, 512, 4096, p_encWh0, 16, 0, 16);
  pack(enc_Wx1, 512, 4096, p_encW1, 32, 0, 16);
  pack(enc_Wh1, 512, 4096, p_encW1, 32, 16, 16);
  pack(enc_Wp0, 1024, 512, p_encWp0, 32, 0, 32);
  pack(enc_Wp1, 1024, 512, p_encWp1, 32, 0, 32);
  pack(dec_Wx0, 256, 4096, p_decWx0, 8, 0, 8);
  pack(dec_Wh0, 512, 4096, p_decWh0, 16, 0, 16);
  pack(dec_Wx1, 512, 4096, p_decW1, 32, 0, 16);
  pack(dec_Wh1, 512, 4096, p_decW1, 32, 16, 16);
  pack(dec_Wp0, 1024, 512, p_decWp0, 32, 0, 32);
  pack(dec_Wp1, 1024, 512, p_decWp1, 32, 0, 32);
  pack(joint_W, 512, 256, p_jW, 16, 0, 16);

  // ---- activations -> padded bf16, then layer-0 input contributions
  cvt(x,    8 * T, 80,  xbf, 96);
  cvt(prev, 8 * U, 256, pbf, 256);
  gemm(xbf, 96,  8 * T, p_encWx0, 3, 4096, enc_b0, xgT_enc, 8);   // T=256 -> shift 8
  gemm(pbf, 256, 8 * U, p_decWx0, 8, 4096, dec_b0, xgT_dec, 6);   // U=64  -> shift 6

  // ---- sequential recurrences: encoder (block 0) || decoder (block 1)
  ChainArgs cEnc{xgT_enc, p_encWh0, p_encWp0, e0_enc, p_encW1, enc_b1, p_encWp1,
                 e1_enc, T};
  ChainArgs cDec{xgT_dec, p_decWh0, p_decWp0, e0_dec, p_decW1, dec_b1, p_decWp1,
                 e1_dec, U};
  lstm_chain_kernel<<<2, 1024, 0, stream>>>(cEnc, cDec);

  // ---- joint, split linearly: E' = e1@Wj + bj ; G' = g1@Wj
  gemm(e1_enc, 512, 8 * T, p_jW, 16, 256, joint_b, Ep, -1);
  gemm(e1_dec, 512, 8 * U, p_jW, 16, 256, nullptr, Gp, -1);

  // ---- fused broadcast-add + tanh over the 134MB output (HBM-bound)
  long long total = (long long)8 * T * U * 256;
  joint_final_kernel<<<(int)((total / 4 + 255) / 256), 256, 0, stream>>>(Ep, Gp, out,
                                                                         T, U, total);
}
